// SimCLR_62045097558541
// MI455X (gfx1250) — matrix operand order repaired and checked by reading, not compiled, not run on hardware
//
#include <hip/hip_runtime.h>
#include <math.h>

// ---------------- problem constants ----------------
#define N_ROWS   8192
#define DIM      512
#define BHALF    4096           // batch half
#define SCALE    10.0f          // 1/T

// ---------------- tiling constants ----------------
#define ROWS_PER_WAVE   16
#define WAVES_PER_BLK   8
#define ROWS_PER_BLK    (ROWS_PER_WAVE * WAVES_PER_BLK)   // 128
#define COL_TILE        128      // columns per LDS-staged tile
#define NTILES          (COL_TILE / 16)                   // 8 wmma tiles / wave
#define KCHUNK          32       // wmma K depth (bf16)
#define COL_SPLIT       8        // grid.y column partitioning

// Use the Tensor Data Mover for LDS staging when the builtin exists.
#if defined(__gfx1250__) && __has_builtin(__builtin_amdgcn_tensor_load_to_lds)
#define USE_TDM 1
#else
#define USE_TDM 0
#endif

typedef __attribute__((ext_vector_type(16))) __bf16  v16bf;
typedef __attribute__((ext_vector_type(8)))  float   v8f;

typedef unsigned int uint32x4 __attribute__((ext_vector_type(4)));
typedef int          int32x8  __attribute__((ext_vector_type(8)));
typedef int          int32x4  __attribute__((ext_vector_type(4)));

union Frag {            // 32 bytes: one 16x32 bf16 A/B fragment per lane
    uint4 q[2];
    v16bf bf;
};

// ---- fp32 <-> bf16 split helpers (RNE) ----
__device__ __forceinline__ unsigned short f2bf(float f) {
    unsigned int u = __float_as_uint(f);
    u += 0x7FFFu + ((u >> 16) & 1u);          // round to nearest even
    return (unsigned short)(u >> 16);
}
__device__ __forceinline__ float bf2f(unsigned short h) {
    return __uint_as_float(((unsigned int)h) << 16);
}

#if USE_TDM
// ---------------------------------------------------------------------
// Issue one TDM 2D tile load: COL_TILE rows x (KCHUNK*2 = 64) bytes,
// row stride DIM*2 bytes, packed contiguously into LDS at lds_byte_addr.
// D# encoding per CDNA5 ISA:
//   group0: count=1, lds_addr, global_addr, type=2
//   group1: data_size=3 (8B units), tile_dim0=8 (64B), tile_dim1=128,
//           tensor_dim0_stride=128 units (1KB), huge tensor dims (no OOB),
//           workgroup_mask=0 (not in a cluster), no pad/iterate/atomic.
//   groups 2/3: zero (2-D tile; tile_dim3/4 = 0 -> unused)
// ---------------------------------------------------------------------
__device__ __forceinline__ void tdm_load_tile(const unsigned short* gsrc,
                                              unsigned lds_byte_addr)
{
    const unsigned long long ga = (unsigned long long)(uintptr_t)gsrc;
    const unsigned td = 1u << 30;             // oversized tensor dims
    uint32x4 g0 = { 1u,                                        // count=1
                    lds_byte_addr,                             // lds_addr
                    (unsigned)(ga & 0xFFFFFFFFu),              // global_addr lo
                    (unsigned)(((ga >> 32) & 0x01FFFFFFu)      // global_addr hi
                               | 0x80000000u) };               // type=2
    int32x8 g1 = { (int)(3u << 16),                            // data_size=8B
                   (int)((td & 0xFFFFu) << 16),                // td0[15:0]<<16
                   (int)((td >> 16) & 0xFFFFu),                // td0 hi | td1 lo<<16
                   (int)(((td >> 16) & 0xFFFFu) | (8u << 16)), // td1 hi | tile_dim0=8
                   (int)COL_TILE,                              // tile_dim1=128, tile_dim2=0
                   (int)(DIM * 2 / 8),                         // dim0 stride = 128 units
                   0, 0 };                                     // stride hi / dim1 stride
    int32x4 gz = { 0, 0, 0, 0 };
#if __clang_major__ >= 23
    int32x8 gz8 = { 0, 0, 0, 0, 0, 0, 0, 0 };
    __builtin_amdgcn_tensor_load_to_lds(g0, g1, gz, gz, gz8, 0);
#else
    __builtin_amdgcn_tensor_load_to_lds(g0, g1, gz, gz, 0);
#endif
}
#endif // USE_TDM

// =====================================================================
// Kernel 1: row-normalize fp32 features and emit bf16 hi/lo split
// one block (256 threads) per row of 512 floats
// =====================================================================
__global__ __launch_bounds__(256)
void normalize_split_kernel(const float* __restrict__ x,
                            unsigned short* __restrict__ hi,
                            unsigned short* __restrict__ lo)
{
    __shared__ float red[8];
    const int row = blockIdx.x;
    const int tid = threadIdx.x;

    float a = x[row * DIM + tid];
    float b = x[row * DIM + tid + 256];
    float ss = a * a + b * b;
    #pragma unroll
    for (int m = 1; m < 32; m <<= 1) ss += __shfl_xor(ss, m, 32);
    if ((tid & 31) == 0) red[tid >> 5] = ss;
    __syncthreads();
    if (tid < 8) {
        float v = red[tid];
        #pragma unroll
        for (int m = 1; m < 8; m <<= 1) v += __shfl_xor(v, m, 32);
        if (tid == 0) red[0] = v;
    }
    __syncthreads();

    float nrm = fmaxf(sqrtf(red[0]), 1e-12f);   // matches reference EPS clamp
    float inv = 1.0f / nrm;

    float va = a * inv, vb = b * inv;
    unsigned short ha = f2bf(va), hb = f2bf(vb);
    hi[row * DIM + tid]       = ha;
    hi[row * DIM + tid + 256] = hb;
    lo[row * DIM + tid]       = f2bf(va - bf2f(ha));
    lo[row * DIM + tid + 256] = f2bf(vb - bf2f(hb));
}

// =====================================================================
// Kernel 2: streaming similarity tiles via bf16-split WMMA.
// grid = (N_ROWS/ROWS_PER_BLK, COL_SPLIT); block = 256 (8 waves).
// Each wave owns 16 rows; 128-column tiles staged in LDS (TDM DMA),
// shared by all 8 waves. Accumulates per-row sum(exp(10*s-10)) (diag
// masked) and the positive logit; deterministic per-split partials.
// =====================================================================
__global__ __launch_bounds__(256)
void simclr_tiles_kernel(const unsigned short* __restrict__ Ahi,
                         const unsigned short* __restrict__ Alo,
                         float* __restrict__ rowsum_part,   // [COL_SPLIT][N_ROWS]
                         float* __restrict__ possum_part)   // [COL_SPLIT][N_ROWS]
{
    __shared__ unsigned short tileB[2][COL_TILE][KCHUNK];   // 16 KB

    const int tid   = threadIdx.x;
    const int wave  = tid >> 5;
    const int lane  = tid & 31;
    const int half  = lane >> 4;       // 0: lanes 0-15, 1: lanes 16-31
    const int lr    = lane & 15;

    const int rowBase  = blockIdx.x * ROWS_PER_BLK + wave * ROWS_PER_WAVE;
    const int colBase  = blockIdx.y * (N_ROWS / COL_SPLIT);
    const int nColTile = (N_ROWS / COL_SPLIT) / COL_TILE;

    const int aRow = rowBase + lr;     // A-fragment row this lane services

#if USE_TDM
    const unsigned ldsHi = (unsigned)(uintptr_t)&tileB[0][0][0];
    const unsigned ldsLo = (unsigned)(uintptr_t)&tileB[1][0][0];
#endif

    float sumAcc[8], posAcc[8];
    #pragma unroll
    for (int v = 0; v < 8; ++v) { sumAcc[v] = 0.0f; posAcc[v] = 0.0f; }

    // Warm WGP$/L2 with this wave's A rows (global_prefetch_b8)
    #pragma unroll 4
    for (int r = 0; r < ROWS_PER_WAVE; ++r) {
        __builtin_prefetch(Ahi + (size_t)(rowBase + r) * DIM, 0, 3);
        __builtin_prefetch(Alo + (size_t)(rowBase + r) * DIM, 0, 3);
    }

    for (int ct = 0; ct < nColTile; ++ct) {
        const int c0 = colBase + ct * COL_TILE;

        v8f acc[NTILES];
        #pragma unroll
        for (int t = 0; t < NTILES; ++t)
            acc[t] = (v8f){0.f,0.f,0.f,0.f,0.f,0.f,0.f,0.f};

        for (int k0 = 0; k0 < DIM; k0 += KCHUNK) {
            // ---- stage B tile chunk (both formats) into LDS ----
            __syncthreads();           // previous chunk fully consumed
#if USE_TDM
            if (wave == 0) {           // uniform branch: one TDM issuer
                tdm_load_tile(Ahi + (size_t)c0 * DIM + k0, ldsHi);
                tdm_load_tile(Alo + (size_t)c0 * DIM + k0, ldsLo);
                __builtin_amdgcn_s_wait_tensorcnt((short)0);
            }
#else
            {
                const int col = tid >> 1;            // 0..127
                const int off = (tid & 1) * 16;      // ushort offset 0 / 16
                const unsigned short* s0 = Ahi + (size_t)(c0 + col) * DIM + k0 + off;
                const unsigned short* s1 = Alo + (size_t)(c0 + col) * DIM + k0 + off;
                *(uint4*)&tileB[0][col][off] = *(const uint4*)s0;
                *(uint4*)&tileB[1][col][off] = *(const uint4*)s1;
            }
#endif
            __syncthreads();

            // ---- A fragments (16x32 bf16 layout, cdna5 CDNA5 ISA) ----
            // lanes 0-15: K = k0+0..7 , k0+16..23 ; lanes 16-31: +8
            const int kbase = k0 + half * 8;
            Frag ah, al;
            ah.q[0] = *(const uint4*)(Ahi + (size_t)aRow * DIM + kbase);
            ah.q[1] = *(const uint4*)(Ahi + (size_t)aRow * DIM + kbase + 16);
            al.q[0] = *(const uint4*)(Alo + (size_t)aRow * DIM + kbase);
            al.q[1] = *(const uint4*)(Alo + (size_t)aRow * DIM + kbase + 16);

            #pragma unroll
            for (int t = 0; t < NTILES; ++t) {
                // B fragment: lane l<16 -> N=l, K=0..7,16..23 ; l>=16 -> N=l-16, K=8..15,24..31
                const int bcol = t * 16 + lr;
                const int bk   = half * 8;
                Frag bh, bl;
                bh.q[0] = *(const uint4*)&tileB[0][bcol][bk];
                bh.q[1] = *(const uint4*)&tileB[0][bcol][bk + 16];
                bl.q[0] = *(const uint4*)&tileB[1][bcol][bk];
                bl.q[1] = *(const uint4*)&tileB[1][bcol][bk + 16];

                // fp32-ish product: hi*hi + hi*lo + lo*hi
                acc[t] = __builtin_amdgcn_wmma_f32_16x16x32_bf16(
                             false, ah.bf, false, bh.bf, (short)0, acc[t], false, false);
                acc[t] = __builtin_amdgcn_wmma_f32_16x16x32_bf16(
                             false, ah.bf, false, bl.bf, (short)0, acc[t], false, false);
                acc[t] = __builtin_amdgcn_wmma_f32_16x16x32_bf16(
                             false, al.bf, false, bh.bf, (short)0, acc[t], false, false);
            }
        }

        // ---- epilogue: exp with fixed shift (unit rows => 10*s <= 10) ----
        // C/D layout: VGPR v -> lanes 0-15 hold (M=v, N=lane),
        //                       lanes 16-31 hold (M=v+8, N=lane-16)
        #pragma unroll
        for (int t = 0; t < NTILES; ++t) {
            const int col = c0 + t * 16 + lr;
            #pragma unroll
            for (int v = 0; v < 8; ++v) {
                const int row = rowBase + v + half * 8;
                const int partner = (row < BHALF) ? row + BHALF : row - BHALF;
                float s = acc[t][v] * SCALE;
                float e = __expf(s - SCALE);
                e = (col == row) ? 0.0f : e;        // mask self-similarity
                sumAcc[v] += e;
                posAcc[v] += (col == partner) ? s : 0.0f;
            }
        }
    }

    // ---- cross-lane row reduction within each 16-lane half ----
    #pragma unroll
    for (int v = 0; v < 8; ++v) {
        float s = sumAcc[v], p = posAcc[v];
        #pragma unroll
        for (int m = 1; m < 16; m <<= 1) {
            s += __shfl_xor(s, m, 32);
            p += __shfl_xor(p, m, 32);
        }
        if (lr == 0) {
            const int row = rowBase + v + half * 8;
            rowsum_part[blockIdx.y * N_ROWS + row] = s;   // unique writer
            possum_part[blockIdx.y * N_ROWS + row] = p;
        }
    }
}

// =====================================================================
// Kernel 3: deterministic final reduction -> scalar loss
// single block of 256 threads
// =====================================================================
__global__ __launch_bounds__(256)
void finalize_kernel(const float* __restrict__ rowsum_part,
                     const float* __restrict__ possum_part,
                     float* __restrict__ out)
{
    __shared__ float red[8];
    float accum = 0.0f;
    for (int r = threadIdx.x; r < N_ROWS; r += 256) {
        float s = 0.0f, p = 0.0f;
        #pragma unroll
        for (int c = 0; c < COL_SPLIT; ++c) {
            s += rowsum_part[c * N_ROWS + r];
            p += possum_part[c * N_ROWS + r];
        }
        // lse = SCALE + log(sum);  contrib = pos - lse ; loss += -contrib
        accum += (SCALE + logf(fmaxf(s, 1e-30f))) - p;
    }
    accum *= (1.0f / (float)BHALF);
    #pragma unroll
    for (int m = 1; m < 32; m <<= 1) accum += __shfl_xor(accum, m, 32);
    if ((threadIdx.x & 31) == 0) red[threadIdx.x >> 5] = accum;
    __syncthreads();
    if (threadIdx.x == 0) {
        float v = 0.0f;
        #pragma unroll
        for (int i = 0; i < 8; ++i) v += red[i];
        out[0] = v;
    }
}

// =====================================================================
// launcher
// workspace layout (bytes):
//   [0)                    rowsum partials  COL_SPLIT*N_ROWS f32 (256 KB)
//   [256KB)                possum partials  COL_SPLIT*N_ROWS f32 (256 KB)
//   [512KB)                Ahi  N_ROWS*DIM  bf16 bits (8 MB)
//   [512KB + 8MB)          Alo  N_ROWS*DIM  bf16 bits (8 MB)
// total ~16.5 MB
// =====================================================================
extern "C" void kernel_launch(void* const* d_in, const int* in_sizes, int n_in,
                              void* d_out, int out_size, void* d_ws, size_t ws_size,
                              hipStream_t stream) {
    (void)in_sizes; (void)n_in; (void)out_size; (void)ws_size;

    const float* features = (const float*)d_in[0];
    float* out = (float*)d_out;

    char* ws = (char*)d_ws;
    float* rowsum_part = (float*)ws;
    float* possum_part = (float*)(ws + (size_t)COL_SPLIT * N_ROWS * sizeof(float));
    unsigned short* Ahi = (unsigned short*)(ws + 2 * (size_t)COL_SPLIT * N_ROWS * sizeof(float));
    unsigned short* Alo = Ahi + (size_t)N_ROWS * DIM;

    normalize_split_kernel<<<N_ROWS, 256, 0, stream>>>(features, Ahi, Alo);

    dim3 grid(N_ROWS / ROWS_PER_BLK, COL_SPLIT);
    simclr_tiles_kernel<<<grid, 256, 0, stream>>>(Ahi, Alo, rowsum_part, possum_part);

    finalize_kernel<<<1, 256, 0, stream>>>(rowsum_part, possum_part, out);
}
